// ContextAwareAttention_7696581394771
// MI455X (gfx1250) — matrix operand order repaired and checked by reading, not compiled, not run on hardware
//
#include <hip/hip_runtime.h>

// ---------------------------------------------------------------------------
// ContextAwareAttention for MI455X (gfx1250, wave32, WMMA bf16)
// Phase 0: elementwise f32 -> bf16 conversion (inputs get +context), weights -> bf16
// Phase 1: three WMMA GEMM projections (pure bf16 load + wmma inner loop);
//          V stored transposed [B,D,S] so the PV GEMM's B-operand is contiguous-in-K.
// Phase 2: flash attention, WMMA bf16 / f32 accum, wave-parallel online softmax,
//          Q tile staged to LDS via gfx1250 async global->LDS DMA.
// ---------------------------------------------------------------------------

typedef __bf16 bf16;
typedef __attribute__((ext_vector_type(16))) __bf16 v16bf;
typedef __attribute__((ext_vector_type(8)))  __bf16 v8bf;
typedef __attribute__((ext_vector_type(8)))  float  v8f;
typedef int v4i __attribute__((vector_size(16)));   // matches async-LDS builtin param

#define WMMA_BF16(a, b, c) \
  __builtin_amdgcn_wmma_f32_16x16x32_bf16(false, (a), false, (b), (short)0, (c), false, false)

constexpr int BATCH = 4;
constexpr int SEQ   = 2048;
constexpr int DIM   = 1024;
constexpr float SCALE = 0.03125f;   // 1/sqrt(1024)

// Probe for gfx1250 async global->LDS path (safe fallback if absent).
#if defined(__has_builtin)
#  if __has_builtin(__builtin_amdgcn_global_load_async_to_lds_b128) && \
      __has_builtin(__builtin_amdgcn_s_wait_asynccnt)
#    define USE_ASYNC_LDS 1
#  endif
#endif
#ifndef USE_ASYNC_LDS
#  define USE_ASYNC_LDS 0
#endif
#define AS1 __attribute__((address_space(1)))
#define AS3 __attribute__((address_space(3)))

// ---------------------------------------------------------------------------
// Phase 0: bf16 conversion, optionally adding broadcast context vector.
// ---------------------------------------------------------------------------
template <bool HAS_CTX>
__global__ __launch_bounds__(256, 1)
void cvt_kernel(const float* __restrict__ x, const float* __restrict__ ctx,
                bf16* __restrict__ out, size_t n)
{
  size_t i = ((size_t)blockIdx.x * 256 + threadIdx.x) * 8;
  if (i >= n) return;
  float4 v0 = *(const float4*)(x + i);
  float4 v1 = *(const float4*)(x + i + 4);
  if (HAS_CTX) {
    int c = (int)(i & (DIM - 1));
    float4 c0 = *(const float4*)(ctx + c);
    float4 c1 = *(const float4*)(ctx + c + 4);
    v0.x += c0.x; v0.y += c0.y; v0.z += c0.z; v0.w += c0.w;
    v1.x += c1.x; v1.y += c1.y; v1.z += c1.z; v1.w += c1.w;
  }
  v8bf o;
  o[0] = (bf16)v0.x; o[1] = (bf16)v0.y; o[2] = (bf16)v0.z; o[3] = (bf16)v0.w;
  o[4] = (bf16)v1.x; o[5] = (bf16)v1.y; o[6] = (bf16)v1.z; o[7] = (bf16)v1.w;
  *(v8bf*)(out + i) = o;
}

// ---------------------------------------------------------------------------
// Phase 1: out[m,n] = sum_k X[m,k]*W[n,k] + bias[n]   (X,W already bf16)
// One 16(M) x 64(N) strip per wave: A fragment reused for 4 WMMAs.
// ---------------------------------------------------------------------------
template <bool TRANSPOSED>
__global__ __launch_bounds__(128, 1)
void proj_kernel(const bf16* __restrict__ Xb, const bf16* __restrict__ Wb,
                 const float* __restrict__ bias,
                 bf16* __restrict__ outRM,   // row-major [B*S, D]
                 bf16* __restrict__ outTR)   // transposed [B, D, S]
{
  const int lane = threadIdx.x & 31;
  const int wave = threadIdx.x >> 5;
  const int hi   = lane >> 4;
  const int lo   = lane & 15;
  const size_t mBase = (size_t)blockIdx.x * 16;            // 512 M-tiles
  const int    nWave = blockIdx.y * 256 + wave * 64;       // 4 N-groups x 4 waves x 64

  v8f acc[4];
#pragma unroll
  for (int t = 0; t < 4; ++t) acc[t] = (v8f){};

  const bf16* aRow = Xb + (mBase + lo) * DIM;
  for (int kb = 0; kb < DIM; kb += 32) {
    v16bf a;
    __builtin_memcpy(&a, aRow + kb + hi * 16, 16);
    __builtin_memcpy((char*)&a + 16, aRow + kb + 8 + hi * 16, 16);
#pragma unroll
    for (int t = 0; t < 4; ++t) {
      const bf16* wp = Wb + (size_t)(nWave + t * 16 + lo) * DIM + kb + hi * 16;
      v16bf bfr;
      __builtin_memcpy(&bfr, wp, 32);
      acc[t] = WMMA_BF16(a, bfr, acc[t]);
    }
  }

#pragma unroll
  for (int t = 0; t < 4; ++t) {
    const int n = nWave + t * 16 + lo;
    const float bval = bias[n];
    if (!TRANSPOSED) {
#pragma unroll
      for (int r = 0; r < 8; ++r) {
        size_t m = mBase + 8 * hi + r;
        outRM[m * DIM + n] = (bf16)(acc[t][r] + bval);
      }
    } else {
      size_t b  = mBase / SEQ;
      size_t s0 = (mBase % SEQ) + 8 * hi;
      v8bf o;
#pragma unroll
      for (int r = 0; r < 8; ++r) o[r] = (bf16)(acc[t][r] + bval);
      *(v8bf*)(outTR + ((b * DIM + n) * (size_t)SEQ + s0)) = o;
    }
  }
}

// ---------------------------------------------------------------------------
// Phase 2: flash attention. 256 threads (8 waves) per (batch, 16 q-rows).
// ---------------------------------------------------------------------------
__global__ __launch_bounds__(256, 1)
void attn_kernel(const bf16* __restrict__ Qb, const bf16* __restrict__ Kb,
                 const bf16* __restrict__ Vt, float* __restrict__ out)
{
  __shared__ __align__(16) bf16  Qs[16][DIM];   // 32 KB
  __shared__ __align__(16) float sS[16][128];   // 8 KB raw scores
  __shared__ __align__(16) bf16  Ps[16][128];   // 4 KB probabilities
  __shared__ float mRun[16], lRun[16], alphaS[16];

  const int tid  = threadIdx.x;
  const int lane = tid & 31;
  const int wave = tid >> 5;
  const int hi   = lane >> 4;
  const int lo   = lane & 15;
  const int qBase = blockIdx.x * 16;
  const int b     = blockIdx.y;

  // ---- Stage Q tile (16 x 1024 bf16 = 2048 x 16B): async DMA to LDS if available
  {
    const bf16* src = Qb + ((size_t)b * SEQ + qBase) * DIM;
#if USE_ASYNC_LDS
    AS1 v4i* g = (AS1 v4i*)(size_t)(const void*)src;
    AS3 v4i* l = (AS3 v4i*)(unsigned)(size_t)(void*)&Qs[0][0];
#pragma unroll
    for (int i = 0; i < 8; ++i) {
      int idx = tid + i * 256;
      __builtin_amdgcn_global_load_async_to_lds_b128(g + idx, l + idx, 0, 0);
    }
    __builtin_amdgcn_s_wait_asynccnt(0);
#else
    const v8bf* s8 = (const v8bf*)src;
    v8bf* d8 = (v8bf*)&Qs[0][0];
#pragma unroll
    for (int i = 0; i < 8; ++i) d8[tid + i * 256] = s8[tid + i * 256];
#endif
  }
  if (tid < 16) { mRun[tid] = -3.0e38f; lRun[tid] = 0.0f; }
  __syncthreads();

  v8f acc[8];
#pragma unroll
  for (int t = 0; t < 8; ++t) acc[t] = (v8f){};

  for (int kt = 0; kt < SEQ / 128; ++kt) {
    const int keyBase = kt * 128;

    // ---- scores: this wave's 16 keys, contract over D (32 WMMAs)
    v8f sAcc = {};
    {
      const int key = keyBase + wave * 16 + lo;
      const bf16* kRow = Kb + ((size_t)b * SEQ + key) * DIM;
      for (int kb = 0; kb < DIM; kb += 32) {
        v16bf a;
        __builtin_memcpy(&a, &Qs[lo][kb + hi * 16], 16);
        __builtin_memcpy((char*)&a + 16, &Qs[lo][kb + 8 + hi * 16], 16);
        v16bf bfr;
        __builtin_memcpy(&bfr, kRow + kb + hi * 16, 32);
        sAcc = WMMA_BF16(a, bfr, sAcc);
      }
    }
    __syncthreads();                      // prev iter's Ps fully consumed
#pragma unroll
    for (int r = 0; r < 8; ++r)
      sS[8 * hi + r][wave * 16 + lo] = sAcc[r] * SCALE;
    __syncthreads();

    // ---- wave-parallel online softmax: half-wave per row, 8 cols/lane,
    //      shfl_xor tree reduction across the 16-lane group.
    {
      const int row = 2 * wave + hi;
      const int c0  = lo * 8;
      float mOld = mRun[row];
      float mNew = mOld;
      float v[8];
#pragma unroll
      for (int i = 0; i < 8; ++i) { v[i] = sS[row][c0 + i]; mNew = fmaxf(mNew, v[i]); }
#pragma unroll
      for (int m = 1; m < 16; m <<= 1) mNew = fmaxf(mNew, __shfl_xor(mNew, m, 32));
      float alpha = __expf(mOld - mNew);
      float sum = 0.0f;
#pragma unroll
      for (int i = 0; i < 8; ++i) {
        float p = __expf(v[i] - mNew);
        Ps[row][c0 + i] = (bf16)p;
        sum += p;
      }
#pragma unroll
      for (int m = 1; m < 16; m <<= 1) sum += __shfl_xor(sum, m, 32);
      if (lo == 0) {
        lRun[row]   = lRun[row] * alpha + sum;
        mRun[row]   = mNew;
        alphaS[row] = alpha;
      }
    }
    __syncthreads();

    // ---- rescale accumulators
#pragma unroll
    for (int t = 0; t < 8; ++t) {
#pragma unroll
      for (int r = 0; r < 8; ++r) acc[t][r] *= alphaS[8 * hi + r];
    }

    // ---- P @ V^T : wave owns output cols [wave*128, +128)  (32 WMMAs)
#pragma unroll
    for (int ks = 0; ks < 4; ++ks) {
      const int kb = ks * 32;
      v16bf a;
      __builtin_memcpy(&a, &Ps[lo][kb + hi * 16], 16);
      __builtin_memcpy((char*)&a + 16, &Ps[lo][kb + 8 + hi * 16], 16);
#pragma unroll
      for (int t = 0; t < 8; ++t) {
        const int dcol = wave * 128 + t * 16 + lo;
        const bf16* vp = Vt + ((size_t)b * DIM + dcol) * SEQ + keyBase + kb + hi * 16;
        v16bf bfr;
        __builtin_memcpy(&bfr, vp, 32);
        acc[t] = WMMA_BF16(a, bfr, acc[t]);
      }
    }
  }

  __syncthreads();
#pragma unroll
  for (int t = 0; t < 8; ++t) {
    const int dcol = wave * 128 + t * 16 + lo;
#pragma unroll
    for (int r = 0; r < 8; ++r) {
      const int q = qBase + 8 * hi + r;
      out[((size_t)b * SEQ + q) * DIM + dcol] = acc[t][r] / lRun[8 * hi + r];
    }
  }
}

// ---------------------------------------------------------------------------
extern "C" void kernel_launch(void* const* d_in, const int* in_sizes, int n_in,
                              void* d_out, int out_size, void* d_ws, size_t ws_size,
                              hipStream_t stream) {
  const float* query   = (const float*)d_in[0];
  const float* key     = (const float*)d_in[1];
  const float* value   = (const float*)d_in[2];
  const float* context = (const float*)d_in[3];
  const float* Wq = (const float*)d_in[4];
  const float* bq = (const float*)d_in[5];
  const float* Wk = (const float*)d_in[6];
  const float* bk = (const float*)d_in[7];
  const float* Wv = (const float*)d_in[8];
  const float* bv = (const float*)d_in[9];
  float* out = (float*)d_out;

  const size_t elems  = (size_t)BATCH * SEQ * DIM;   // 8M
  const size_t welems = (size_t)DIM * DIM;           // 1M
  bf16* Qb  = (bf16*)d_ws;          // projected Q  (16 MB)
  bf16* Kb  = Qb  + elems;          // projected K  (16 MB)
  bf16* Vt  = Kb  + elems;          // projected V^T (16 MB)
  bf16* Xq  = Vt  + elems;          // (query+ctx) bf16
  bf16* Xk  = Xq  + elems;          // (key+ctx) bf16
  bf16* Xv  = Xk  + elems;          // value bf16
  bf16* Wqb = Xv  + elems;          // weights bf16 (2 MB each)
  bf16* Wkb = Wqb + welems;
  bf16* Wvb = Wkb + welems;

  // Phase 0: conversions
  const int cvtBlk = 256;
  dim3 cg((unsigned)(elems / (8 * cvtBlk)));         // 4096 blocks
  dim3 wg((unsigned)(welems / (8 * cvtBlk)));        // 512 blocks
  cvt_kernel<true ><<<cg, cvtBlk, 0, stream>>>(query, context, Xq, elems);
  cvt_kernel<true ><<<cg, cvtBlk, 0, stream>>>(key,   context, Xk, elems);
  cvt_kernel<false><<<cg, cvtBlk, 0, stream>>>(value, nullptr, Xv, elems);
  cvt_kernel<false><<<wg, cvtBlk, 0, stream>>>(Wq, nullptr, Wqb, welems);
  cvt_kernel<false><<<wg, cvtBlk, 0, stream>>>(Wk, nullptr, Wkb, welems);
  cvt_kernel<false><<<wg, cvtBlk, 0, stream>>>(Wv, nullptr, Wvb, welems);

  // Phase 1: projections (grid: 512 M-tiles x 4 N-groups of 256)
  dim3 pg(BATCH * SEQ / 16, DIM / 256, 1);
  proj_kernel<false><<<pg, 128, 0, stream>>>(Xq, Wqb, bq, Qb, nullptr);
  proj_kernel<false><<<pg, 128, 0, stream>>>(Xk, Wkb, bk, Kb, nullptr);
  proj_kernel<true ><<<pg, 128, 0, stream>>>(Xv, Wvb, bv, nullptr, Vt);

  // Phase 2: attention
  dim3 ag(SEQ / 16, BATCH, 1);
  attn_kernel<<<ag, 256, 0, stream>>>(Qb, Kb, Vt, out);
}
